// GATEdgeClassifier_13211319402837
// MI455X (gfx1250) — matrix operand order repaired and checked by reading, not compiled, not run on hardware
//
#include <hip/hip_runtime.h>
#include <hip/hip_bf16.h>

typedef __attribute__((ext_vector_type(16))) _Float16 v16h;
typedef __attribute__((ext_vector_type(8)))  float    v8f;
typedef __attribute__((ext_vector_type(4)))  _Float16 v4h;

// ---- problem constants (match reference) ----
#define NN     10000          // nodes (625 * 16)
#define NE     160000         // edges (10000 * 16)
#define EPP    (NE + NN)      // edges + self loops
#define NEGS   0.2f
#define NATT   4
#define CEDGE  32768          // MLP edge chunk (16-aligned; last chunk 28928 = 1808*16)

// ---------------- device helpers ----------------
__device__ __forceinline__ unsigned fkey(float x) {
  unsigned u = __float_as_uint(x);
  return (u >> 31) ? ~u : (u | 0x80000000u);   // monotonic f32 -> u32
}
__device__ __forceinline__ float unfkey(unsigned k) {
  unsigned u = (k >> 31) ? (k ^ 0x80000000u) : ~k;
  return __uint_as_float(u);
}
__device__ __forceinline__ float eluf(float x)   { return x > 0.f ? x : (__expf(x) - 1.f); }
__device__ __forceinline__ float lreluf(float x) { return x > 0.f ? x : NEGS * x; }

// ---------------- WMMA GEMM: C[M,Nc] f32 = A[M,K] f32 * Bp (fragment-packed f16) ----
// Requirements: M % 16 == 0, K % 32 == 0. block = 128 threads (4 waves).
// Bp layout: tile (tk,tn) -> 512 halves: lane*16 + j, where
//   element (j,lane): B[tk*32 + ((lane&16)?16:0) + j][tn*16 + (lane&15)]
__global__ __launch_bounds__(128)
void gemm_a32_bp(const float* __restrict__ A, const _Float16* __restrict__ Bp,
                 float* __restrict__ C, int M, int K, int Nc, int ntn)
{
  __shared__ _Float16 As[16][32];
  const int tid  = threadIdx.x;
  const int lane = tid & 31, wv = tid >> 5;
  const int mBase = blockIdx.x * 16;
  const int tn    = blockIdx.y * 4 + wv;
  const int m     = lane & 15;
  const int abase = (lane < 16) ? 0 : 8;
  const int n     = lane & 15;
  // A staging: thread -> one float4 of row r, cols c4..c4+3
  const int sr = tid >> 3, sc4 = (tid & 7) << 2;
  const float* aptr = A + (size_t)(mBase + sr) * K + sc4;
  v8f acc = {};
  for (int kt = 0, tk = 0; kt < K; kt += 32, ++tk) {
    __syncthreads();
    {
      const float4 f = *(const float4*)(aptr + kt);
      v4h h;
      h[0] = (_Float16)f.x; h[1] = (_Float16)f.y;
      h[2] = (_Float16)f.z; h[3] = (_Float16)f.w;
      *(v4h*)(&As[sr][sc4]) = h;
      if (kt + 32 < K) __builtin_prefetch(aptr + kt + 32, 0, 3);
    }
    __syncthreads();
    v16h a, b = {};
#pragma unroll
    for (int j = 0; j < 16; ++j) a[j] = As[m][2 * abase + j];
    if (tn < ntn) b = *(const v16h*)(Bp + ((size_t)tk * ntn + tn) * 512 + lane * 16);
    acc = __builtin_amdgcn_wmma_f32_16x16x32_f16(false, a, false, b, (short)0, acc,
                                                 false, false);
  }
  const int col = tn * 16 + n;
  if (col < Nc) {
    const int rb = mBase + ((lane < 16) ? 0 : 8);
#pragma unroll
    for (int r = 0; r < 8; ++r) C[(size_t)(rb + r) * Nc + col] = acc[r];
  }
}

// WMMA GEMM over gathered edge rows: er[e] = concat(xh[row[e]], xh[col[e]]); K=128, Nc=64.
// eCnt % 16 == 0. Bp fragment-packed (ntn = 4).
__global__ __launch_bounds__(128)
void gemm_er(const float* __restrict__ xh, const int* __restrict__ row,
             const int* __restrict__ col, const _Float16* __restrict__ Bp,
             float* __restrict__ C, int eBase)
{
  __shared__ _Float16 As[16][128];
  const int tid = threadIdx.x;
  // stage 16 edges x 128 features as float4 chunks: 512 float4 / 128 threads = 4 each
#pragma unroll
  for (int t = 0; t < 4; ++t) {
    int idx = tid + t * 128;
    int r = idx >> 5, c4 = (idx & 31) << 2;         // c4 in {0,4,...,124}
    int e = eBase + blockIdx.x * 16 + r;
    int nd = (c4 < 64) ? row[e] : col[e];
    const float4 f = *(const float4*)(xh + (size_t)nd * 64 + (c4 & 63));
    v4h h;
    h[0] = (_Float16)f.x; h[1] = (_Float16)f.y;
    h[2] = (_Float16)f.z; h[3] = (_Float16)f.w;
    *(v4h*)(&As[r][c4]) = h;
  }
  __syncthreads();
  const int lane = tid & 31, wv = tid >> 5;
  const int m = lane & 15, abase = (lane < 16) ? 0 : 8, n = lane & 15;
  v8f acc = {};
#pragma unroll
  for (int tk = 0; tk < 4; ++tk) {
    v16h a, b;
#pragma unroll
    for (int j = 0; j < 16; ++j) a[j] = As[m][tk * 32 + 2 * abase + j];
    b = *(const v16h*)(Bp + ((size_t)tk * 4 + wv) * 512 + lane * 16);
    acc = __builtin_amdgcn_wmma_f32_16x16x32_f16(false, a, false, b, (short)0, acc,
                                                 false, false);
  }
  const int rb = blockIdx.x * 16 + ((lane < 16) ? 0 : 8);
#pragma unroll
  for (int r = 0; r < 8; ++r) C[(size_t)(rb + r) * 64 + wv * 16 + n] = acc[r];
}

// ---------------- weight packing ----------------
// Generic: W f32 [K][Nc] -> fragment-packed f16 tiles; total = (K/32)*ntn*512 halves
__global__ void k_pack_b(const float* __restrict__ W, _Float16* __restrict__ Bp,
                         int K, int Nc, int ntn) {
  int g = blockIdx.x * blockDim.x + threadIdx.x;
  int total = (K >> 5) * ntn * 512;
  if (g >= total) return;
  int tile = g >> 9, r = g & 511, lane = r >> 4, j = r & 15;
  int tk = tile / ntn, tn = tile - tk * ntn;
  int kk = tk * 32 + (lane & 16) + j;
  int cc = tn * 16 + (lane & 15);
  float v = (cc < Nc) ? W[(size_t)kk * Nc + cc] : 0.f;
  Bp[g] = (_Float16)v;
}

// NNConv reshuffle + pack: logical W2r[i][col] with
//   col<256: enn_w2[(col>>5)*1024 + i*32 + (col&31)] ; col>=256: enn_b2[i*32 + col-256]
// K=32, Nc=288, ntn=18 -> 18*512 halves
__global__ void k_pack_w2r(const float* __restrict__ w2, const float* __restrict__ b2,
                           _Float16* __restrict__ Bp) {
  int g = blockIdx.x * blockDim.x + threadIdx.x;
  if (g >= 18 * 512) return;
  int tn = g >> 9, r = g & 511, lane = r >> 4, j = r & 15;
  int i  = (lane & 16) + j;          // K index (input channel)
  int cc = tn * 16 + (lane & 15);    // 0..287
  float v;
  if (cc < 256) { int k = cc >> 5, o = cc & 31; v = w2[k * 1024 + i * 32 + o]; }
  else          { v = b2[i * 32 + (cc - 256)]; }
  Bp[g] = (_Float16)v;
}

// ---------------- small / irregular kernels ----------------
__global__ void k_edge_nn1(const float* __restrict__ ea, const float* __restrict__ w1,
                           const float* __restrict__ b1, float* __restrict__ h) {
  int g = blockIdx.x * blockDim.x + threadIdx.x;
  if (g >= NE * 8) return;
  int e = g >> 3, k = g & 7;
  float a0 = ea[e * 3 + 0], a1 = ea[e * 3 + 1], a2 = ea[e * 3 + 2];
  float v = a0 * w1[k] + a1 * w1[8 + k] + a2 * w1[16 + k] + b1[k];
  h[g] = v > 0.f ? v : 0.f;
}

__global__ void k_nnconv_scatter(const float* __restrict__ h, const float* __restrict__ g,
                                 const int* __restrict__ row, const int* __restrict__ col,
                                 float* __restrict__ xh0) {
  int gid = blockIdx.x * blockDim.x + threadIdx.x;
  if (gid >= NE * 32) return;
  int e = gid >> 5, o = gid & 31;
  const float* gr = g + (size_t)row[e] * 288;
  const float* he = h + (size_t)e * 8;
  float acc = gr[256 + o];                       // enn_b2 contribution
#pragma unroll
  for (int k = 0; k < 8; ++k) acc += he[k] * gr[k * 32 + o];
  atomicAdd(xh0 + (size_t)col[e] * 32 + o, acc);
}

__global__ void k_nnconv_root(const float* __restrict__ x, const float* __restrict__ rw,
                              const float* __restrict__ bias, float* __restrict__ xh0) {
  int gid = blockIdx.x * blockDim.x + threadIdx.x;
  if (gid >= NN * 32) return;
  int n = gid >> 5, o = gid & 31;
  float acc = xh0[gid] + bias[o];
#pragma unroll
  for (int i = 0; i < 32; ++i) acc += x[(size_t)n * 32 + i] * rw[i * 32 + o];
  xh0[gid] = eluf(acc);
}

__global__ void k_att_scores(const float* __restrict__ xw, const float* __restrict__ as,
                             const float* __restrict__ ad, float* __restrict__ ss,
                             float* __restrict__ sd) {
  int gid = blockIdx.x * blockDim.x + threadIdx.x;
  if (gid >= NN * 8) return;
  int n = gid >> 3, h = gid & 7;
  const float* xr = xw + (size_t)n * 512 + h * 64;
  float s1 = 0.f, s2 = 0.f;
#pragma unroll 8
  for (int c = 0; c < 64; ++c) { s1 += xr[c] * as[h * 64 + c]; s2 += xr[c] * ad[h * 64 + c]; }
  ss[gid] = s1; sd[gid] = s2;
}

__global__ void k_edge_max(const int* __restrict__ row, const int* __restrict__ col,
                           const float* __restrict__ ss, const float* __restrict__ sd,
                           unsigned* __restrict__ mkey) {
  int gid = blockIdx.x * blockDim.x + threadIdx.x;
  if (gid >= EPP * 8) return;
  int e = gid >> 3, h = gid & 7;
  int s, d;
  if (e < NE) { s = row[e]; d = col[e]; } else { s = d = e - NE; }
  float el = lreluf(ss[s * 8 + h] + sd[d * 8 + h]);
  atomicMax(&mkey[d * 8 + h], fkey(el));
}

__global__ void k_edge_exp(const int* __restrict__ row, const int* __restrict__ col,
                           const float* __restrict__ ss, const float* __restrict__ sd,
                           const unsigned* __restrict__ mkey, float* __restrict__ p,
                           float* __restrict__ den) {
  int gid = blockIdx.x * blockDim.x + threadIdx.x;
  if (gid >= EPP * 8) return;
  int e = gid >> 3, h = gid & 7;
  int s, d;
  if (e < NE) { s = row[e]; d = col[e]; } else { s = d = e - NE; }
  float el = lreluf(ss[s * 8 + h] + sd[d * 8 + h]);
  float pv = __expf(el - unfkey(mkey[d * 8 + h]));
  p[gid] = pv;
  atomicAdd(&den[d * 8 + h], pv);
}

__global__ void k_edge_aggr(const int* __restrict__ row, const int* __restrict__ col,
                            const float* __restrict__ xw, const float* __restrict__ p,
                            const float* __restrict__ den, float* __restrict__ agg) {
  long long gid = (long long)blockIdx.x * blockDim.x + threadIdx.x;
  if (gid >= (long long)EPP * 512) return;
  int e = (int)(gid >> 9), ch = (int)(gid & 511), h = ch >> 6;
  int s, d;
  if (e < NE) { s = row[e]; d = col[e]; } else { s = d = e - NE; }
  float alpha = p[e * 8 + h] / den[d * 8 + h];
  atomicAdd(agg + (size_t)d * 512 + ch, alpha * xw[(size_t)s * 512 + ch]);
}

__global__ void k_gat_final(const float* __restrict__ agg, const float* __restrict__ bias,
                            float* __restrict__ xh, int residual) {
  int gid = blockIdx.x * blockDim.x + threadIdx.x;
  if (gid >= NN * 64) return;
  int n = gid >> 6, c = gid & 63;
  float v = 0.f;
#pragma unroll
  for (int h = 0; h < 8; ++h) v += agg[(size_t)n * 512 + h * 64 + c];
  v = v * 0.125f + bias[c];
  if (residual) v += xh[gid];
  xh[gid] = eluf(v);
}

__global__ void k_relu_bias(float* __restrict__ h, const float* __restrict__ b, int count) {
  int gid = blockIdx.x * blockDim.x + threadIdx.x;
  if (gid >= count) return;
  float v = h[gid] + b[gid & 63];
  h[gid] = v > 0.f ? v : 0.f;
}

__global__ void k_mlp_final(const float* __restrict__ t2, const float* __restrict__ h1,
                            const float* __restrict__ b2, const float* __restrict__ w3,
                            const float* __restrict__ b3, float* __restrict__ out,
                            int eBase, int eCnt) {
  int le = blockIdx.x * blockDim.x + threadIdx.x;
  if (le >= eCnt) return;
  const float* t  = t2 + (size_t)le * 64;
  const float* hh = h1 + (size_t)le * 64;
  float acc = b3[0];
#pragma unroll 8
  for (int c = 0; c < 64; ++c) {
    float v = t[c] + b2[c] + hh[c];
    v = v > 0.f ? v : 0.f;
    acc += v * w3[c];
  }
  out[eBase + le] = acc;
}

// ---------------- host launch ----------------
extern "C" void kernel_launch(void* const* d_in, const int* in_sizes, int n_in,
                              void* d_out, int out_size, void* d_ws, size_t ws_size,
                              hipStream_t stream) {
  (void)in_sizes; (void)n_in; (void)out_size; (void)ws_size;
  const float* x        = (const float*)d_in[0];
  const int*   ei       = (const int*)d_in[1];
  const int*   row      = ei;
  const int*   col      = ei + NE;
  const float* eattr    = (const float*)d_in[2];
  const float* enn_w1   = (const float*)d_in[3];
  const float* enn_b1   = (const float*)d_in[4];
  const float* enn_w2   = (const float*)d_in[5];
  const float* enn_b2   = (const float*)d_in[6];
  const float* root_w   = (const float*)d_in[7];
  const float* nc_bias  = (const float*)d_in[8];
  const float* gat0_w   = (const float*)d_in[9];
  const float* gat_w    = (const float*)d_in[10];
  const float* att_src  = (const float*)d_in[11];
  const float* att_dst  = (const float*)d_in[12];
  const float* gat_bias = (const float*)d_in[13];
  const float* mlp_w1   = (const float*)d_in[14];
  const float* mlp_b1   = (const float*)d_in[15];
  const float* mlp_w2   = (const float*)d_in[16];
  const float* mlp_b2   = (const float*)d_in[17];
  const float* mlp_w3   = (const float*)d_in[18];
  const float* mlp_b3   = (const float*)d_in[19];
  float* out = (float*)d_out;

  // ---- workspace layout: persistent + one reused scratch region (~52 MB) ----
  char* ws = (char*)d_ws;
  size_t off = 0;
  auto take = [&](size_t bytes) { size_t o = off; off = (off + bytes + 255) & ~(size_t)255; return o; };
  size_t oW2R = take((size_t)18 * 512 * 2);           // packed W2r (K=32, ntn=18)
  size_t oG0P = take((size_t)1 * 32 * 512 * 2);       // packed gat0_w (K=32, ntn=32)
  size_t oGWP = take((size_t)3 * 2 * 32 * 512 * 2);   // packed gat_w x3 (K=64, ntn=32)
  size_t oM1P = take((size_t)4 * 4 * 512 * 2);        // packed mlp_w1 (K=128, ntn=4)
  size_t oM2P = take((size_t)2 * 4 * 512 * 2);        // packed mlp_w2 (K=64, ntn=4)
  size_t oXH0 = take((size_t)NN * 32 * 4);
  size_t oXH  = take((size_t)NN * 64 * 4);
  size_t oSS  = take((size_t)NN * 8 * 4);
  size_t oSD  = take((size_t)NN * 8 * 4);
  size_t oMK  = take((size_t)NN * 8 * 4);
  size_t oDEN = take((size_t)NN * 8 * 4);
  size_t oP   = take((size_t)EPP * 8 * 4);
  size_t oSCR = take(2 * (size_t)NN * 512 * 4);       // max phase: GAT (xw + agg)

  _Float16* w2rp = (_Float16*)(ws + oW2R);
  _Float16* g0p  = (_Float16*)(ws + oG0P);
  _Float16* gwp  = (_Float16*)(ws + oGWP);
  _Float16* m1p  = (_Float16*)(ws + oM1P);
  _Float16* m2p  = (_Float16*)(ws + oM2P);
  float* xh0 = (float*)(ws + oXH0);
  float* xh  = (float*)(ws + oXH);
  float* ss  = (float*)(ws + oSS);
  float* sd  = (float*)(ws + oSD);
  unsigned* mkey = (unsigned*)(ws + oMK);
  float* den = (float*)(ws + oDEN);
  float* p   = (float*)(ws + oP);
  // phase overlays inside scratch
  float* he  = (float*)(ws + oSCR);                                               // E x 8
  float* gbf = (float*)(ws + oSCR + (((size_t)NE * 8 * 4 + 255) & ~(size_t)255)); // N x 288
  float* xw  = (float*)(ws + oSCR);                                               // N x 512
  float* agg = (float*)(ws + oSCR + (size_t)NN * 512 * 4);                        // N x 512
  float* b1c = (float*)(ws + oSCR);                                               // CE x 64
  float* b2c = (float*)(ws + oSCR + (size_t)CEDGE * 64 * 4);

  const int TB = 256;
  auto blocks = [](long long n, int tb) { return (unsigned)((n + tb - 1) / tb); };

  // 1) weight repack into WMMA fragment order
  k_pack_b<<<blocks(32 * 512, TB), TB, 0, stream>>>(gat0_w, g0p, 32, 512, 32);
  for (int i = 0; i < 3; ++i)
    k_pack_b<<<blocks(2 * 32 * 512, TB), TB, 0, stream>>>(
        gat_w + (size_t)i * 64 * 512, gwp + (size_t)i * 2 * 32 * 512, 64, 512, 32);
  k_pack_b<<<blocks(4 * 4 * 512, TB), TB, 0, stream>>>(mlp_w1, m1p, 128, 64, 4);
  k_pack_b<<<blocks(2 * 4 * 512, TB), TB, 0, stream>>>(mlp_w2, m2p, 64, 64, 4);
  k_pack_w2r<<<blocks(18 * 512, TB), TB, 0, stream>>>(enn_w2, enn_b2, w2rp);

  // 2) NNConv (restructured): g = x @ W2r (WMMA), per-edge 8-MAC apply + scatter, root+elu
  k_edge_nn1<<<blocks((long long)NE * 8, TB), TB, 0, stream>>>(eattr, enn_w1, enn_b1, he);
  gemm_a32_bp<<<dim3(NN / 16, 5), 128, 0, stream>>>(x, w2rp, gbf, NN, 32, 288, 18);
  hipMemsetAsync(xh0, 0, (size_t)NN * 32 * 4, stream);
  k_nnconv_scatter<<<blocks((long long)NE * 32, TB), TB, 0, stream>>>(he, gbf, row, col, xh0);
  k_nnconv_root<<<blocks((long long)NN * 32, TB), TB, 0, stream>>>(x, root_w, nc_bias, xh0);

  // 3) GAT stack (WMMA transform + atomic segment softmax + atomic aggregation)
  for (int i = 0; i < NATT; ++i) {
    if (i == 0)
      gemm_a32_bp<<<dim3(NN / 16, 8), 128, 0, stream>>>(xh0, g0p, xw, NN, 32, 512, 32);
    else
      gemm_a32_bp<<<dim3(NN / 16, 8), 128, 0, stream>>>(
          xh, gwp + (size_t)(i - 1) * 2 * 32 * 512, xw, NN, 64, 512, 32);
    k_att_scores<<<blocks((long long)NN * 8, TB), TB, 0, stream>>>(
        xw, att_src + (size_t)i * 512, att_dst + (size_t)i * 512, ss, sd);
    hipMemsetAsync(mkey, 0, (size_t)NN * 8 * 4, stream);
    hipMemsetAsync(den, 0, (size_t)NN * 8 * 4, stream);
    hipMemsetAsync(agg, 0, (size_t)NN * 512 * 4, stream);
    k_edge_max<<<blocks((long long)EPP * 8, TB), TB, 0, stream>>>(row, col, ss, sd, mkey);
    k_edge_exp<<<blocks((long long)EPP * 8, TB), TB, 0, stream>>>(row, col, ss, sd, mkey, p, den);
    k_edge_aggr<<<blocks((long long)EPP * 512, TB), TB, 0, stream>>>(row, col, xw, p, den, agg);
    k_gat_final<<<blocks((long long)NN * 64, TB), TB, 0, stream>>>(
        agg, gat_bias + (size_t)i * 64, xh, i > 0 ? 1 : 0);
  }

  // 4) edge MLP, chunked (WMMA for both dense layers)
  for (int eb = 0; eb < NE; eb += CEDGE) {
    int ec = (NE - eb < CEDGE) ? (NE - eb) : CEDGE;
    gemm_er<<<dim3(ec / 16), 128, 0, stream>>>(xh, row, col, m1p, b1c, eb);
    k_relu_bias<<<blocks((long long)ec * 64, TB), TB, 0, stream>>>(b1c, mlp_b1, ec * 64);
    gemm_a32_bp<<<dim3(ec / 16, 1), 128, 0, stream>>>(b1c, m2p, b2c, ec, 64, 64, 4);
    k_mlp_final<<<blocks(ec, TB), TB, 0, stream>>>(b2c, b1c, mlp_b2, mlp_w3, mlp_b3, out, eb, ec);
  }
}
